// _RNNLayer_9947144257683
// MI455X (gfx1250) — matrix operand order repaired and checked by reading, not compiled, not run on hardware
//
#include <hip/hip_runtime.h>
#include <math.h>

// Problem sizes (fixed by the reference)
#define S_LEN 1024
#define B_DIM 64
#define I_DIM 512
#define H_DIM 512

typedef __attribute__((ext_vector_type(16))) __bf16 v16bf;
typedef __attribute__((ext_vector_type(2)))  __bf16 v2bf;
typedef __attribute__((ext_vector_type(8)))  float  v8f;
typedef __attribute__((ext_vector_type(4)))  unsigned int v4u;
typedef __attribute__((ext_vector_type(8)))  int v8i;
typedef __attribute__((ext_vector_type(4)))  int v4i;

#if __has_builtin(__builtin_amdgcn_tensor_load_to_lds) && __has_builtin(__builtin_amdgcn_s_wait_tensorcnt)
#define HAVE_TDM 1
#else
#define HAVE_TDM 0
#endif

// ---------- helpers ----------

__device__ __forceinline__ float fast_tanh(float x) {
#if __has_builtin(__builtin_amdgcn_tanhf)
    return __builtin_amdgcn_tanhf(x);     // single v_tanh_f32 trans op
#else
    return tanhf(x);
#endif
}

__device__ __forceinline__ unsigned pk2(float a, float b) {
#if __has_builtin(__builtin_amdgcn_cvt_pk_bf16_f32)
    v2bf r = __builtin_amdgcn_cvt_pk_bf16_f32(a, b);
    return __builtin_bit_cast(unsigned, r);
#else
    __bf16 x = (__bf16)a, y = (__bf16)b;
    unsigned short ux = __builtin_bit_cast(unsigned short, x);
    unsigned short uy = __builtin_bit_cast(unsigned short, y);
    return (unsigned)ux | ((unsigned)uy << 16);
#endif
}

__device__ __forceinline__ uint2 pack4(float4 f) {
    return make_uint2(pk2(f.x, f.y), pk2(f.z, f.w));
}

// Convert two 8-float contiguous chunks into a 16-element bf16 fragment
__device__ __forceinline__ v16bf cvt_frag(const float* p0, const float* p1) {
    union { unsigned u[8]; v16bf v; } r;
#pragma unroll
    for (int i = 0; i < 4; ++i) r.u[i]     = pk2(p0[2 * i], p0[2 * i + 1]);
#pragma unroll
    for (int i = 0; i < 4; ++i) r.u[4 + i] = pk2(p1[2 * i], p1[2 * i + 1]);
    return r.v;
}

// A-matrix fragment (16x32 bf16): lanes 0-15 hold K{0..7,16..23}, lanes 16-31 K{8..15,24..31}
__device__ __forceinline__ v16bf load_frag_a(const __bf16* rowp, int koff) {
    union { uint4 q[2]; v16bf v; } u;
    u.q[0] = *(const uint4*)(rowp + koff);
    u.q[1] = *(const uint4*)(rowp + koff + 16);
    return u.v;
}

// B-matrix fragment (32x16 bf16): lanes 0-15 hold K{0..7,16..23}, lanes 16-31 K{8..15,24..31}
__device__ __forceinline__ v16bf load_frag_b(const __bf16* rowp, int koff) {
    union { uint4 q[2]; v16bf v; } u;
    u.q[0] = *(const uint4*)(rowp + koff);
    u.q[1] = *(const uint4*)(rowp + koff + 16);
    return u.v;
}

__device__ __forceinline__ v8f wmma_bf16(v16bf a, v16bf b, v8f c) {
    return __builtin_amdgcn_wmma_f32_16x16x32_bf16(
        false, a, false, b, (short)0, c, false, false);
}

#if HAVE_TDM
// Issue one 2D TDM tile load: global (row-major, stride0 elements/row) -> LDS.
// Descriptor per CDNA5 ISA ch.8: group0 = {count/lds/global/type}, group1 = {sizes/pads/dims}.
__device__ __forceinline__ void tdm_load_2d(unsigned lds_off, unsigned long long gaddr,
                                            int dsz_log2, int dim0, int dim1, long stride0,
                                            int tile0, int tile1,
                                            int pad_interval, int pad_amount) {
    v4u g0;
    g0[0] = 1u;                                              // count=1, user mode
    g0[1] = lds_off;                                         // LDS byte address
    g0[2] = (unsigned)(gaddr & 0xffffffffull);               // global_addr[31:0]
    g0[3] = (unsigned)((gaddr >> 32) & 0x01ffffffull) | 0x80000000u; // addr[56:32] | type=2
    v8i g1;
    g1[0] = (dsz_log2 << 16) | (1 << 20) | (pad_interval << 22) | (pad_amount << 25);
    g1[1] = (dim0 & 0xffff) << 16;                           // tensor_dim0[15:0]
    g1[2] = ((dim0 >> 16) & 0xffff) | ((dim1 & 0xffff) << 16);
    g1[3] = ((dim1 >> 16) & 0xffff) | ((tile0 & 0xffff) << 16);
    g1[4] = (tile1 & 0xffff);                                // tile_dim1; tile_dim2=0
    g1[5] = (int)(stride0 & 0xffffffff);                     // tensor_dim0_stride[31:0]
    g1[6] = (int)((stride0 >> 32) & 0xffff);
    g1[7] = 0;
    v4i z4 = {0, 0, 0, 0};
#if defined(__clang_major__) && __clang_major__ >= 23
    v8i z8 = {0, 0, 0, 0, 0, 0, 0, 0};
    __builtin_amdgcn_tensor_load_to_lds(g0, g1, z4, z4, z8, 0);
#else
    __builtin_amdgcn_tensor_load_to_lds(g0, g1, z4, z4, 0);
#endif
}
#endif

// Fallback staging: 64x512 bf16 tile global -> LDS (row stride 520)
__device__ __forceinline__ void stage_tile_bf16(const __bf16* __restrict__ g,
                                                __bf16* __restrict__ s, int tid) {
#pragma unroll 4
    for (int i = 0; i < 16; ++i) {
        int idx = tid + i * 256;          // 4096 16B chunks
        int row = idx >> 6;
        int c8  = (idx & 63) * 8;
        *(uint4*)&s[row * 520 + c8] = *(const uint4*)&g[row * 512 + c8];
    }
}

// ---------- kernel 0: prep (Wh -> bf16, zero h0 ping buffer, zero barrier counters) ----------

__global__ __launch_bounds__(256) void prep_kernel(const float* __restrict__ wh_fw,
                                                   const float* __restrict__ wh_bw,
                                                   __bf16* __restrict__ whbf,
                                                   unsigned* __restrict__ hzero,
                                                   unsigned* __restrict__ cnt) {
    int gid = blockIdx.x * 256 + threadIdx.x;     // grid 512x256 = 131072
    const float4* src = (gid < 65536) ? (const float4*)wh_fw : (const float4*)wh_bw;
    float4 f = src[gid & 65535];
    ((uint2*)whbf)[gid] = pack4(f);
    if (gid < 32768) hzero[gid] = 0u;             // zero ping buffer 0 (both dirs, bf16)
    if (gid < 2) cnt[gid] = 0u;
}

// ---------- kernel 1: x_proj = x @ Wi^T + bi  (both directions) ----------
// M = S*B = 65536, N = 512/dir, K = 512. WG tile 64x128, 8 waves (4M x 2N), 4 C tiles/wave.

__global__ __launch_bounds__(256) void xproj_kernel(
        const float* __restrict__ x,
        const float* __restrict__ wi_fw, const float* __restrict__ bi_fw,
        const float* __restrict__ wi_bw, const float* __restrict__ bi_bw,
        float* __restrict__ out) {
    const int bid = blockIdx.x;        // 8192 blocks
    const int nt  = bid & 3;
    const int mt  = (bid >> 2) & 1023;
    const int d   = bid >> 12;
    const float* wi = d ? wi_bw : wi_fw;
    const float* bi = d ? bi_bw : bi_fw;

    const int tid  = threadIdx.x;
    const int lane = tid & 31;
    const int wv   = tid >> 5;
    const int mw   = wv & 3;
    const int nw   = wv >> 2;
    const int m_base = mt * 64;
    const int n_base = nt * 128;

    v8f acc[4];
#pragma unroll
    for (int j = 0; j < 4; ++j)
#pragma unroll
        for (int r = 0; r < 8; ++r) acc[j][r] = 0.0f;

    const int koffA = (lane < 16) ? 0 : 8;
    const int koffB = (lane < 16) ? 0 : 8;
    const int arow  = 16 * mw + (lane & 15);

#if HAVE_TDM
    // f32 tiles DMA'd straight into LDS by the TDM, double-buffered; TDM padding
    // gives row stride 36 f32 (conflict-free b128 reads). Convert f32->bf16 in consumer.
    __shared__ __align__(16) float sAf[2][64 * 36];
    __shared__ __align__(16) float sBf[2][128 * 36];
    const float* xtile = x  + (size_t)m_base * I_DIM;
    const float* wtile = wi + (size_t)n_base * I_DIM;

    if (wv == 0) {
        tdm_load_2d((unsigned)(size_t)&sAf[0][0], (unsigned long long)(size_t)xtile,
                    2, I_DIM, 64, I_DIM, 32, 64, /*pad_int 32DW*/4, /*pad 4DW*/3);
        tdm_load_2d((unsigned)(size_t)&sBf[0][0], (unsigned long long)(size_t)wtile,
                    2, I_DIM, 128, I_DIM, 32, 128, 4, 3);
    }
    for (int kc = 0; kc < 16; ++kc) {
        const int cur = kc & 1;
        if (wv == 0) {
            if (kc < 15) {
                tdm_load_2d((unsigned)(size_t)&sAf[cur ^ 1][0],
                            (unsigned long long)(size_t)(xtile + (kc + 1) * 32),
                            2, I_DIM, 64, I_DIM, 32, 64, 4, 3);
                tdm_load_2d((unsigned)(size_t)&sBf[cur ^ 1][0],
                            (unsigned long long)(size_t)(wtile + (kc + 1) * 32),
                            2, I_DIM, 128, I_DIM, 32, 128, 4, 3);
                __builtin_amdgcn_s_wait_tensorcnt((short)2);   // current tiles done
            } else {
                __builtin_amdgcn_s_wait_tensorcnt((short)0);
            }
        }
        __syncthreads();
        const float* Ar = &sAf[cur][arow * 36];
        v16bf a = cvt_frag(Ar + koffA, Ar + koffA + 16);
#pragma unroll
        for (int j = 0; j < 4; ++j) {
            const float* Br = &sBf[cur][(nw * 64 + j * 16 + (lane & 15)) * 36];
            v16bf b = cvt_frag(Br + koffB, Br + koffB + 16);
            acc[j] = wmma_bf16(a, b, acc[j]);
        }
        __syncthreads();
    }
#else
    __shared__ __bf16 sA[64 * 40];
    __shared__ __bf16 sB[128 * 40];
    for (int kc = 0; kc < 16; ++kc) {
        const int k0 = kc * 32;
#pragma unroll
        for (int i = 0; i < 2; ++i) {
            int idx = tid + i * 256;
            int row = idx >> 3;
            int kq  = (idx & 7) * 4;
            float4 f = *(const float4*)(x + (size_t)(m_base + row) * I_DIM + k0 + kq);
            *(uint2*)&sA[row * 40 + kq] = pack4(f);
        }
#pragma unroll
        for (int i = 0; i < 4; ++i) {
            int idx = tid + i * 256;
            int row = idx >> 3;
            int kq  = (idx & 7) * 4;
            float4 f = *(const float4*)(wi + (size_t)(n_base + row) * I_DIM + k0 + kq);
            *(uint2*)&sB[row * 40 + kq] = pack4(f);
        }
        __syncthreads();
        v16bf a = load_frag_a(&sA[arow * 40], koffA);
#pragma unroll
        for (int j = 0; j < 4; ++j) {
            int brow = nw * 64 + j * 16 + (lane & 15);
            v16bf b = load_frag_b(&sB[brow * 40], koffB);
            acc[j] = wmma_bf16(a, b, acc[j]);
        }
        __syncthreads();
    }
#endif

    // epilogue: + bias, store f32 into out[S,B,2H]
    const int mrow0 = m_base + 16 * mw + ((lane >> 4) ? 8 : 0);
#pragma unroll
    for (int j = 0; j < 4; ++j) {
        int ncol = n_base + nw * 64 + j * 16 + (lane & 15);
        float bv = bi[ncol];
#pragma unroll
        for (int r = 0; r < 8; ++r) {
            out[(size_t)(mrow0 + r) * (2 * H_DIM) + (size_t)d * H_DIM + ncol] = acc[j][r] + bv;
        }
    }
}

// ---------- kernel 2: persistent recurrence ----------
// 16 WGs: d = bid/8, slice = bid%8 (64 cols of H). Wh slice resident in LDS (bf16,
// TDM-loaded once). h kept bf16, ping-ponged through L2; each step = one TDM load.

__global__ __launch_bounds__(256) void rnn_step_kernel(
        const __bf16* __restrict__ whbf,
        const float* __restrict__ bh_fw, const float* __restrict__ bh_bw,
        float* __restrict__ out, __bf16* __restrict__ hb16, unsigned* __restrict__ cnt) {
    __shared__ __align__(16) __bf16 sW[64 * 520];
    __shared__ __align__(16) __bf16 sH[64 * 520];

    const int bid   = blockIdx.x;     // 16 blocks
    const int d     = bid >> 3;
    const int slice = bid & 7;
    const int n_base = slice * 64;
    const float* bh = d ? bh_bw : bh_fw;

    const int tid  = threadIdx.x;
    const int lane = tid & 31;
    const int wv   = tid >> 5;
    const int mi0 = (2 * wv) & 3;
    const int mi1 = mi0 + 1;
    const int ni  = (2 * wv) >> 2;

    const __bf16* wtile = whbf + ((size_t)d * H_DIM + n_base) * H_DIM;
#if HAVE_TDM
    if (wv == 0)
        tdm_load_2d((unsigned)(size_t)&sW[0], (unsigned long long)(size_t)wtile,
                    1, H_DIM, 64, H_DIM, H_DIM, 64, /*pad_int 256DW*/7, /*pad 4DW*/3);
#else
    stage_tile_bf16(wtile, sW, tid);
#endif

    const int koffA = (lane < 16) ? 0 : 8;
    const int koffB = (lane < 16) ? 0 : 8;
    const int arow0 = 16 * mi0 + (lane & 15);
    const int arow1 = 16 * mi1 + (lane & 15);
    const int brow  = 16 * ni  + (lane & 15);
    const int mrowL0 = 16 * mi0 + ((lane >> 4) ? 8 : 0);
    const int mrowL1 = 16 * mi1 + ((lane >> 4) ? 8 : 0);
    const int ng = n_base + 16 * ni + (lane & 15);
    const float bhv = bh[ng];

    unsigned* c = cnt + d;

    for (int t = 0; t < S_LEN; ++t) {
        const int s = d ? (S_LEN - 1 - t) : t;
        const __bf16* hprev = hb16 + ((size_t)(t & 1) * 2 + d) * (B_DIM * H_DIM);
        __bf16*       hnext = hb16 + ((size_t)((t + 1) & 1) * 2 + d) * (B_DIM * H_DIM);

#if HAVE_TDM
        if (wv == 0) {
            tdm_load_2d((unsigned)(size_t)&sH[0], (unsigned long long)(size_t)hprev,
                        1, H_DIM, B_DIM, H_DIM, H_DIM, B_DIM, 7, 3);
            __builtin_amdgcn_s_wait_tensorcnt((short)0);   // also covers the sW load
        }
#else
        stage_tile_bf16(hprev, sH, tid);
#endif
        __syncthreads();

        // init accumulators from x_proj (stored in out)
        const size_t outbase = (size_t)s * B_DIM * (2 * H_DIM) + (size_t)d * H_DIM + ng;
        v8f a0, a1;
#pragma unroll
        for (int r = 0; r < 8; ++r) {
            a0[r] = out[outbase + (size_t)(mrowL0 + r) * (2 * H_DIM)];
            a1[r] = out[outbase + (size_t)(mrowL1 + r) * (2 * H_DIM)];
        }

        // h_{t-1} @ Wh^T over K = 512 (two C tiles share the B fragment)
#pragma unroll 4
        for (int kc = 0; kc < 16; ++kc) {
            const int k0 = kc * 32;
            v16bf bf  = load_frag_b(&sW[brow * 520 + k0], koffB);
            v16bf af0 = load_frag_a(&sH[arow0 * 520 + k0], koffA);
            v16bf af1 = load_frag_a(&sH[arow1 * 520 + k0], koffA);
            a0 = wmma_bf16(af0, bf, a0);
            a1 = wmma_bf16(af1, bf, a1);
        }

        // epilogue: hw-tanh, write out (in place over xp), write bf16 h, h_n at final step
#pragma unroll
        for (int r = 0; r < 8; ++r) {
            float h0 = fast_tanh(a0[r] + bhv);
            float h1 = fast_tanh(a1[r] + bhv);
            out[outbase + (size_t)(mrowL0 + r) * (2 * H_DIM)] = h0;
            out[outbase + (size_t)(mrowL1 + r) * (2 * H_DIM)] = h1;
            hnext[(mrowL0 + r) * H_DIM + ng] = (__bf16)h0;
            hnext[(mrowL1 + r) * H_DIM + ng] = (__bf16)h1;
            if (t == S_LEN - 1) {
                float* hn = out + (size_t)S_LEN * B_DIM * (2 * H_DIM) + (size_t)d * B_DIM * H_DIM;
                hn[(mrowL0 + r) * H_DIM + ng] = h0;
                hn[(mrowL1 + r) * H_DIM + ng] = h1;
            }
        }

        // prefetch next step's x_proj rows while we wait at the barrier
        if (t + 1 < S_LEN) {
            const int s2 = d ? (s - 1) : (s + 1);
            const size_t nb = (size_t)s2 * B_DIM * (2 * H_DIM) + (size_t)d * H_DIM + ng;
            __builtin_prefetch(&out[nb + (size_t)mrowL0 * (2 * H_DIM)], 0, 1);
            __builtin_prefetch(&out[nb + (size_t)mrowL1 * (2 * H_DIM)], 0, 1);
        }

        // device-scope release, then per-direction global barrier (8 WGs)
        __threadfence();
        __syncthreads();
        if (tid == 0) {
            atomicAdd(c, 1u);
            const unsigned target = 8u * (unsigned)(t + 1);
            while (__hip_atomic_load(c, __ATOMIC_ACQUIRE, __HIP_MEMORY_SCOPE_AGENT) < target) {
                __builtin_amdgcn_s_sleep(1);
            }
        }
        __syncthreads();
    }
}

// ---------- host launcher ----------

extern "C" void kernel_launch(void* const* d_in, const int* in_sizes, int n_in,
                              void* d_out, int out_size, void* d_ws, size_t ws_size,
                              hipStream_t stream) {
    const float* x     = (const float*)d_in[0];
    const float* wi_fw = (const float*)d_in[1];
    const float* wh_fw = (const float*)d_in[2];
    const float* bi_fw = (const float*)d_in[3];
    const float* bh_fw = (const float*)d_in[4];
    const float* wi_bw = (const float*)d_in[5];
    const float* wh_bw = (const float*)d_in[6];
    const float* bi_bw = (const float*)d_in[7];
    const float* bh_bw = (const float*)d_in[8];
    float* out = (float*)d_out;

    // ws layout: whbf (2x512x512 bf16, 1MB) | hb16 (2pp x 2dir x 64x512 bf16, 256KB) | cnt
    __bf16*   whbf = (__bf16*)d_ws;
    __bf16*   hb16 = (__bf16*)((char*)d_ws + 524288);
    unsigned* cnt  = (unsigned*)((char*)d_ws + 524288 + 262144);

    hipLaunchKernelGGL(prep_kernel, dim3(512), dim3(256), 0, stream,
                       wh_fw, wh_bw, whbf, (unsigned*)hb16, cnt);
    hipLaunchKernelGGL(xproj_kernel, dim3(8192), dim3(256), 0, stream,
                       x, wi_fw, bi_fw, wi_bw, bi_bw, out);
    hipLaunchKernelGGL(rnn_step_kernel, dim3(16), dim3(256), 0, stream,
                       whbf, bh_fw, bh_bw, out, hb16, cnt);
}
